// SO2EquivariantGraphAttention_22660247454484
// MI455X (gfx1250) — matrix operand order repaired and checked by reading, not compiled, not run on hardware
//
#include <hip/hip_runtime.h>
#include <hip/hip_bf16.h>

#define E_EDGES 20000
#define NNODES  1000

typedef __attribute__((ext_vector_type(16))) __bf16 v16bf;
typedef __attribute__((ext_vector_type(8)))  float  v8f;

__constant__ int c_PERM[19]   = {0,2,6,11,16, 3,7,12,17, 1,5,10,15, 8,13,18, 4,9,14};
__constant__ int c_EXPAND[25] = {0,1,1,1,2,2,2,2,2,3,3,3,3,3,3,3,4,4,4,4,4,4,4,4,4};

__device__ __forceinline__ unsigned short f2bf(float f) {
  unsigned int u = __float_as_uint(f);
  u += 0x7FFFu + ((u >> 16) & 1u);          // round-to-nearest-even
  return (unsigned short)(u >> 16);
}
__device__ __forceinline__ float siluf(float x) { return x / (1.f + __expf(-x)); }
__device__ __forceinline__ unsigned fenc(float f) {
  unsigned u = __float_as_uint(f);
  return (u & 0x80000000u) ? ~u : (u | 0x80000000u);
}
__device__ __forceinline__ float fdec(unsigned u) {
  return (u & 0x80000000u) ? __uint_as_float(u & 0x7FFFFFFFu) : __uint_as_float(~u);
}

// ---------------------------------------------------------------- f32 -> bf16
__global__ void k_f32_to_bf16(const float* __restrict__ in,
                              unsigned short* __restrict__ out, int n) {
  int i = blockIdx.x * 256 + threadIdx.x;
  if (i < n) out[i] = f2bf(in[i]);
}

// ---------------------------------------------------------------- init accums
__global__ void k_init(float* __restrict__ node, float* __restrict__ den,
                       unsigned* __restrict__ amax) {
  int i = blockIdx.x * 256 + threadIdx.x;
  if (i < NNODES * 3200) node[i] = 0.f;
  if (i < NNODES * 8) { den[i] = 0.f; amax[i] = 0u; }
}

// ---------------------------------------------------------------- radial MLP
__global__ __launch_bounds__(64) void k_radial(
    const float* __restrict__ xd,
    const float* __restrict__ w1, const float* __restrict__ b1,
    const float* __restrict__ g1, const float* __restrict__ be1,
    const float* __restrict__ w2, const float* __restrict__ b2,
    const float* __restrict__ g2, const float* __restrict__ be2,
    const float* __restrict__ w3, const float* __restrict__ b3,
    float* __restrict__ rad) {
  __shared__ float sx[64], sh[64], red[64];
  int e = blockIdx.x, t = threadIdx.x;
  sx[t] = xd[(long long)e * 64 + t];
  __syncthreads();
  // layer 1
  float acc = b1[t];
  for (int k = 0; k < 64; ++k) acc += sx[k] * w1[t * 64 + k];
  red[t] = acc; __syncthreads();
  for (int s = 32; s > 0; s >>= 1) { if (t < s) red[t] += red[t + s]; __syncthreads(); }
  float mu = red[0] * (1.f / 64.f); __syncthreads();
  float d = acc - mu;
  red[t] = d * d; __syncthreads();
  for (int s = 32; s > 0; s >>= 1) { if (t < s) red[t] += red[t + s]; __syncthreads(); }
  float var = red[0] * (1.f / 64.f); __syncthreads();
  sh[t] = siluf(d * rsqrtf(var + 1e-5f) * g1[t] + be1[t]);
  __syncthreads();
  // layer 2
  acc = b2[t];
  for (int k = 0; k < 64; ++k) acc += sh[k] * w2[t * 64 + k];
  red[t] = acc; __syncthreads();
  for (int s = 32; s > 0; s >>= 1) { if (t < s) red[t] += red[t + s]; __syncthreads(); }
  mu = red[0] * (1.f / 64.f); __syncthreads();
  d = acc - mu;
  red[t] = d * d; __syncthreads();
  for (int s = 32; s > 0; s >>= 1) { if (t < s) red[t] += red[t + s]; __syncthreads(); }
  var = red[0] * (1.f / 64.f); __syncthreads();
  float y2 = siluf(d * rsqrtf(var + 1e-5f) * g2[t] + be2[t]);
  __syncthreads();
  sh[t] = y2;
  __syncthreads();
  // layer 3: 1536 outputs
  for (int o = 0; o < 24; ++o) {
    int n = o * 64 + t;
    float a3 = b3[n];
    for (int k = 0; k < 64; ++k) a3 += sh[k] * w3[n * 64 + k];
    rad[(long long)e * 1536 + n] = a3;
  }
}

// ------------------------------------------- wigner rotate + rad scale + bf16 stage1
__global__ __launch_bounds__(128) void k_wigner_stage1(
    const float* __restrict__ x, const float* __restrict__ wig,
    const float* __restrict__ rad, const int* __restrict__ eidx,
    unsigned short* __restrict__ stage1) {
  __shared__ float sw[475];      // 19 x 25
  __shared__ float sm[19 * 128]; // rotated msg, perm not yet applied
  int e = blockIdx.x, t = threadIdx.x;
  for (int i = t; i < 475; i += 128) sw[i] = wig[(long long)e * 475 + i];
  __syncthreads();
  int src = eidx[e], dst = eidx[E_EDGES + e];
  const float* xp = (t < 64) ? (x + (long long)src * 1600 + t)
                             : (x + (long long)dst * 1600 + (t - 64));
  float xc[25];
#pragma unroll
  for (int j = 0; j < 25; ++j) xc[j] = xp[j * 64];
  for (int i = 0; i < 19; ++i) {
    float a = 0.f;
#pragma unroll
    for (int j = 0; j < 25; ++j) a += sw[i * 25 + j] * xc[j];
    sm[i * 128 + t] = a;
  }
  __syncthreads();
  const float* re = rad + (long long)e * 1536;
  unsigned short* sp = stage1 + (long long)e * 2432;
  int c = t;
#pragma unroll
  for (int r = 0; r < 5; ++r)
    sp[r * 128 + c] = f2bf(sm[c_PERM[r] * 128 + c] * re[r * 128 + c]);
#pragma unroll
  for (int r = 0; r < 4; ++r) {
    float rv = re[640 + r * 128 + c];
    sp[640 + r * 128 + c]  = f2bf(sm[c_PERM[5 + r] * 128 + c] * rv);
    sp[1152 + r * 128 + c] = f2bf(sm[c_PERM[9 + r] * 128 + c] * rv);
  }
#pragma unroll
  for (int r = 0; r < 3; ++r) {
    float rv = re[1152 + r * 128 + c];
    sp[1664 + r * 128 + c] = f2bf(sm[c_PERM[13 + r] * 128 + c] * rv);
    sp[2048 + r * 128 + c] = f2bf(sm[c_PERM[16 + r] * 128 + c] * rv);
  }
}

// ---------------------------------------------------------------- WMMA GEMM
// C[M,N] (f32) = A[M,K] (bf16, SO2-staged rows) x W[N,K]^T (bf16) + bias
union ABpack { unsigned int u[8]; v16bf v; };

__global__ __launch_bounds__(256) void k_gemm_bf16(
    const unsigned short* __restrict__ A, long long edgeStride,
    long long partStride, int pairMode,
    const unsigned short* __restrict__ B, const float* __restrict__ bias,
    float* __restrict__ C, int M, int N, int K) {
  const int wave = threadIdx.x >> 5;
  const int lane = threadIdx.x & 31;
  const int lm = lane & 15;
  const int lh = lane >> 4;
  const int n0 = blockIdx.x * 64;
  const int m0 = blockIdx.y * 128 + wave * 16;

  v8f acc[4];
#pragma unroll
  for (int s = 0; s < 4; ++s)
#pragma unroll
    for (int i = 0; i < 8; ++i) acc[s][i] = 0.f;

  int mrow = m0 + lm;
  if (mrow > M - 1) mrow = M - 1;
  const unsigned short* Arow =
      A + (pairMode ? ((long long)(mrow >> 1) * edgeStride +
                       (long long)(mrow & 1) * partStride)
                    : (long long)mrow * edgeStride);

  for (int k0 = 0; k0 < K; k0 += 32) {
    ABpack a;
#pragma unroll
    for (int v = 0; v < 8; ++v) {
      int kk = ((v < 4) ? 0 : 16) + (lh ? 8 : 0) + 2 * (v & 3); // ISA 16-bit A 16x32 layout
      a.u[v] = *(const unsigned int*)(Arow + k0 + kk);
    }
#pragma unroll
    for (int s = 0; s < 4; ++s) {
      ABpack b;
      const unsigned short* Bcol =
          B + (long long)(n0 + s * 16 + lm) * K + k0;
#pragma unroll
      for (int v = 0; v < 8; ++v) b.u[v] = *(const unsigned int*)(Bcol + ((v < 4) ? 0 : 16) + (lh ? 8 : 0) + 2 * (v & 3));
      acc[s] = __builtin_amdgcn_wmma_f32_16x16x32_bf16(
          false, a.v, false, b.v, (short)0, acc[s], false, false);
    }
  }

#pragma unroll
  for (int s = 0; s < 4; ++s) {
    int n = n0 + s * 16 + lm;
    float bv = bias ? bias[n] : 0.f;
#pragma unroll
    for (int v = 0; v < 8; ++v) {
      int m = m0 + lh * 8 + v;
      if (m < M) C[(long long)m * N + n] = acc[s][v] + bv;
    }
  }
}

// ------------------------------------------- conv1 real/imag recombine -> msg_hid (perm)
__global__ __launch_bounds__(256) void k_combine1(
    const float* __restrict__ Y0, const float* __restrict__ Y1,
    const float* __restrict__ Y2, float* __restrict__ msgh) {
  int e = blockIdx.x, t = threadIdx.x;
  const float* y1a = Y1 + (long long)(2 * e) * 512;
  const float* y1b = Y1 + (long long)(2 * e + 1) * 512;
  const float* y2a = Y2 + (long long)(2 * e) * 384;
  const float* y2b = Y2 + (long long)(2 * e + 1) * 384;
  for (int idx = t; idx < 1216; idx += 256) {
    int j = idx >> 6, c = idx & 63;
    float v;
    if (j < 5)       v = Y0[(long long)e * 896 + 576 + j * 64 + c];
    else if (j < 9)  { int r = j - 5;  v = y1a[r * 64 + c] - y1b[256 + r * 64 + c]; }
    else if (j < 13) { int r = j - 9;  v = y1b[r * 64 + c] + y1a[256 + r * 64 + c]; }
    else if (j < 16) { int r = j - 13; v = y2a[r * 64 + c] - y2b[192 + r * 64 + c]; }
    else             { int r = j - 16; v = y2b[r * 64 + c] + y2a[192 + r * 64 + c]; }
    msgh[(long long)e * 1216 + idx] = v;
  }
}

// ------------------------------------------- S2 grid nonlinearity + gate -> bf16 stage2
__global__ __launch_bounds__(64) void k_grid_stage2(
    const float* __restrict__ msgh, const float* __restrict__ Y0,
    const float* __restrict__ to_grid, const float* __restrict__ from_grid,
    unsigned short* __restrict__ stage2) {
  __shared__ float s_tg[1900], s_fg[1900], s_m[1216];
  int e = blockIdx.x, t = threadIdx.x;
  for (int i = t; i < 1900; i += 64) {
    int b = i / 19, j = i % 19;
    s_tg[i] = to_grid[b * 19 + c_PERM[j]];
    s_fg[i] = from_grid[b * 19 + c_PERM[j]];
  }
  for (int i = t; i < 1216; i += 64) s_m[i] = msgh[(long long)e * 1216 + i];
  __syncthreads();
  float acc[19];
#pragma unroll
  for (int j = 0; j < 19; ++j) acc[j] = 0.f;
  for (int b = 0; b < 100; ++b) {
    float g = 0.f;
#pragma unroll
    for (int j = 0; j < 19; ++j) g += s_tg[b * 19 + j] * s_m[j * 64 + t];
    g = siluf(g);
#pragma unroll
    for (int j = 0; j < 19; ++j) acc[j] += s_fg[b * 19 + j] * g;
  }
  unsigned short* sp = stage2 + (long long)e * 1216;
  sp[t] = f2bf(siluf(Y0[(long long)e * 896 + 512 + t]));   // perm row 0 = silu(gate)
#pragma unroll
  for (int j = 1; j < 5; ++j)  sp[j * 64 + t] = f2bf(acc[j]);
#pragma unroll
  for (int j = 5; j < 9; ++j)  sp[320 + (j - 5) * 64 + t] = f2bf(acc[j]);
#pragma unroll
  for (int j = 9; j < 13; ++j) sp[576 + (j - 9) * 64 + t] = f2bf(acc[j]);
#pragma unroll
  for (int j = 13; j < 16; ++j) sp[768 + (j - 13) * 64 + t] = f2bf(acc[j]);
#pragma unroll
  for (int j = 16; j < 19; ++j) sp[960 + (j - 16) * 64 + t] = f2bf(acc[j]);
}

// ------------------------------------------- conv2 recombine -> msg2 (perm, E x 19 x 128)
__global__ __launch_bounds__(256) void k_combine2(
    const float* __restrict__ Z0, const float* __restrict__ Z1,
    const float* __restrict__ Z2, float* __restrict__ msg2) {
  int e = blockIdx.x, t = threadIdx.x;
  const float* z1a = Z1 + (long long)(2 * e) * 1024;
  const float* z1b = Z1 + (long long)(2 * e + 1) * 1024;
  const float* z2a = Z2 + (long long)(2 * e) * 768;
  const float* z2b = Z2 + (long long)(2 * e + 1) * 768;
  for (int idx = t; idx < 2432; idx += 256) {
    int j = idx >> 7, c = idx & 127;
    float v;
    if (j < 5)       v = Z0[(long long)e * 640 + j * 128 + c];
    else if (j < 9)  { int r = j - 5;  v = z1a[r * 128 + c] - z1b[512 + r * 128 + c]; }
    else if (j < 13) { int r = j - 9;  v = z1b[r * 128 + c] + z1a[512 + r * 128 + c]; }
    else if (j < 16) { int r = j - 13; v = z2a[r * 128 + c] - z2b[384 + r * 128 + c]; }
    else             { int r = j - 16; v = z2b[r * 128 + c] + z2a[384 + r * 128 + c]; }
    msg2[(long long)e * 2432 + idx] = v;
  }
}

// ---------------------------------------------------------------- attention logits
__global__ __launch_bounds__(256) void k_alpha1(
    const float* __restrict__ Y0, const float* __restrict__ g,
    const float* __restrict__ b, const float* __restrict__ adot,
    const int* __restrict__ eidx, float* __restrict__ alpha,
    unsigned* __restrict__ amax) {
  int gid = blockIdx.x * 256 + threadIdx.x;
  if (gid >= E_EDGES * 8) return;
  int e = gid >> 3, h = gid & 7;
  const float* xr = Y0 + (long long)e * 896 + h * 64;
  float s = 0.f, ss = 0.f;
  for (int k = 0; k < 64; ++k) { float v = xr[k]; s += v; ss += v * v; }
  float mu = s * (1.f / 64.f);
  float var = ss * (1.f / 64.f) - mu * mu;
  float inv = rsqrtf(var + 1e-5f);
  float acc = 0.f;
  for (int k = 0; k < 64; ++k) {
    float v = (xr[k] - mu) * inv * g[k] + b[k];
    float sl = 0.6f * v + 0.4f * v * (2.f / (1.f + __expf(-v)) - 1.f);
    acc += sl * adot[h * 64 + k];
  }
  alpha[gid] = acc;
  int dst = eidx[E_EDGES + e];
  atomicMax(&amax[dst * 8 + h], fenc(acc));
}

__global__ __launch_bounds__(256) void k_alpha2(
    float* __restrict__ alpha, const unsigned* __restrict__ amax,
    float* __restrict__ den, const int* __restrict__ eidx) {
  int gid = blockIdx.x * 256 + threadIdx.x;
  if (gid >= E_EDGES * 8) return;
  int e = gid >> 3, h = gid & 7;
  int dst = eidx[E_EDGES + e];
  float ex = __expf(alpha[gid] - fdec(amax[dst * 8 + h]));
  alpha[gid] = ex;
  atomicAdd(&den[dst * 8 + h], ex);
}

// ------------------------------- alpha-weight, wigner-inverse rotate, segment-sum
__global__ __launch_bounds__(128) void k_attn_scatter(
    const float* __restrict__ msg2, const float* __restrict__ wig_inv,
    const float* __restrict__ alpha, const float* __restrict__ den,
    const int* __restrict__ eidx, float* __restrict__ node) {
  __shared__ float s_wi[475]; // 25 x 19, columns gathered via PERM
  int e = blockIdx.x, t = threadIdx.x;
  int dst = eidx[E_EDGES + e];
  for (int i = t; i < 475; i += 128) {
    int row = i / 19, j = i % 19;
    s_wi[i] = wig_inv[(long long)e * 475 + row * 19 + c_PERM[j]];
  }
  __syncthreads();
  int h = t >> 4;
  float a = alpha[e * 8 + h] / (den[dst * 8 + h] + 1e-16f);
  float m[19];
#pragma unroll
  for (int j = 0; j < 19; ++j) m[j] = msg2[(long long)e * 2432 + j * 128 + t] * a;
  for (int i = 0; i < 25; ++i) {
    float v = 0.f;
#pragma unroll
    for (int j = 0; j < 19; ++j) v += s_wi[i * 19 + j] * m[j];
    atomicAdd(&node[(long long)dst * 3200 + i * 128 + t], v);
  }
}

// ---------------------------------------------------------------- output projection
__global__ __launch_bounds__(256) void k_out_proj(
    const float* __restrict__ node, const float* __restrict__ proj_w,
    const float* __restrict__ proj_b, float* __restrict__ out) {
  __shared__ float s_n[3200];
  int n = blockIdx.x, t = threadIdx.x;
  for (int i = t; i < 3200; i += 256) s_n[i] = node[(long long)n * 3200 + i];
  __syncthreads();
  for (int idx = t; idx < 1600; idx += 256) {
    int m = idx >> 6, o = idx & 63;
    const float* w = proj_w + (long long)c_EXPAND[m] * 8192 + o * 128;
    float acc = (m == 0) ? proj_b[o] : 0.f;
    for (int c = 0; c < 128; ++c) acc += s_n[m * 128 + c] * w[c];
    out[(long long)n * 1600 + idx] = acc;
  }
}

// ================================================================ host launcher
extern "C" void kernel_launch(void* const* d_in, const int* in_sizes, int n_in,
                              void* d_out, int out_size, void* d_ws, size_t ws_size,
                              hipStream_t stream) {
  (void)in_sizes; (void)n_in; (void)out_size; (void)ws_size;
  const float* x        = (const float*)d_in[0];
  const float* edist    = (const float*)d_in[1];
  const int*   eidx     = (const int*)d_in[2];
  const float* wigner   = (const float*)d_in[3];
  const float* wig_inv  = (const float*)d_in[4];
  const float* to_grid  = (const float*)d_in[5];
  const float* from_grid= (const float*)d_in[6];
  const float* rad_w1 = (const float*)d_in[7];
  const float* rad_b1 = (const float*)d_in[8];
  const float* rad_g1 = (const float*)d_in[9];
  const float* rad_be1= (const float*)d_in[10];
  const float* rad_w2 = (const float*)d_in[11];
  const float* rad_b2 = (const float*)d_in[12];
  const float* rad_g2 = (const float*)d_in[13];
  const float* rad_be2= (const float*)d_in[14];
  const float* rad_w3 = (const float*)d_in[15];
  const float* rad_b3 = (const float*)d_in[16];
  const float* c1_w0  = (const float*)d_in[17];
  const float* c1_b0  = (const float*)d_in[18];
  const float* c1_w1  = (const float*)d_in[19];
  const float* c1_w2  = (const float*)d_in[20];
  const float* c2_w0  = (const float*)d_in[21];
  const float* c2_b0  = (const float*)d_in[22];
  const float* c2_w1  = (const float*)d_in[23];
  const float* c2_w2  = (const float*)d_in[24];
  const float* aln_g  = (const float*)d_in[25];
  const float* aln_b  = (const float*)d_in[26];
  const float* adot   = (const float*)d_in[27];
  const float* proj_w = (const float*)d_in[28];
  const float* proj_b = (const float*)d_in[29];

  size_t off = 0;
  auto take = [&](size_t bytes) -> void* {
    void* p = (char*)d_ws + off;
    off += (bytes + 255) & ~(size_t)255;
    return p;
  };
  const size_t E = E_EDGES;
  float* regA = (float*)take(E * 2432 * 4);            // rad (E x 1536), later msg2 (E x 2432)
  unsigned short* regB = (unsigned short*)take(E * 2432 * 2); // stage1 (E x 2432), later stage2 (E x 1216)
  float* Y0   = (float*)take(E * 896 * 4);             // conv1 m0 out + extra (alpha source)
  float* regD = (float*)take((size_t)40000 * 1024 * 4); // Y1 (40000x512) then Z1 (40000x1024)
  float* regE = (float*)take((size_t)40000 * 768 * 4);  // Y2 (40000x384) then Z2 (40000x768)
  float* regF = (float*)take(E * 1216 * 4);            // msg_hid (E x 1216) then Z0 (E x 640)
  unsigned short* w1b = (unsigned short*)take((size_t)896 * 640 * 2);
  unsigned short* w2b = (unsigned short*)take((size_t)512 * 512 * 2);
  unsigned short* w3b = (unsigned short*)take((size_t)384 * 384 * 2);
  unsigned short* w4b = (unsigned short*)take((size_t)640 * 320 * 2);
  unsigned short* w5b = (unsigned short*)take((size_t)1024 * 256 * 2);
  unsigned short* w6b = (unsigned short*)take((size_t)768 * 192 * 2);
  float* alpha = (float*)take(E * 8 * 4);
  unsigned* amax = (unsigned*)take((size_t)NNODES * 8 * 4);
  float* den  = (float*)take((size_t)NNODES * 8 * 4);
  float* node = (float*)take((size_t)NNODES * 3200 * 4);

  float* rad  = regA;  float* msg2 = regA;
  unsigned short* stage1 = regB; unsigned short* stage2 = regB;
  float* Y1 = regD; float* Z1 = regD;
  float* Y2 = regE; float* Z2 = regE;
  float* msgh = regF; float* Z0 = regF;

  // weights -> bf16
  k_f32_to_bf16<<<(896 * 640 + 255) / 256, 256, 0, stream>>>(c1_w0, w1b, 896 * 640);
  k_f32_to_bf16<<<(512 * 512 + 255) / 256, 256, 0, stream>>>(c1_w1, w2b, 512 * 512);
  k_f32_to_bf16<<<(384 * 384 + 255) / 256, 256, 0, stream>>>(c1_w2, w3b, 384 * 384);
  k_f32_to_bf16<<<(640 * 320 + 255) / 256, 256, 0, stream>>>(c2_w0, w4b, 640 * 320);
  k_f32_to_bf16<<<(1024 * 256 + 255) / 256, 256, 0, stream>>>(c2_w1, w5b, 1024 * 256);
  k_f32_to_bf16<<<(768 * 192 + 255) / 256, 256, 0, stream>>>(c2_w2, w6b, 768 * 192);

  k_init<<<(NNODES * 3200 + 255) / 256, 256, 0, stream>>>(node, den, amax);

  // radial MLP + edge-message staging
  k_radial<<<E_EDGES, 64, 0, stream>>>(edist, rad_w1, rad_b1, rad_g1, rad_be1,
                                       rad_w2, rad_b2, rad_g2, rad_be2,
                                       rad_w3, rad_b3, rad);
  k_wigner_stage1<<<E_EDGES, 128, 0, stream>>>(x, wigner, rad, eidx, stage1);

  // SO2 conv1 (WMMA bf16)
  { dim3 g(896 / 64, (20000 + 127) / 128);
    k_gemm_bf16<<<g, 256, 0, stream>>>(stage1, 2432, 0, 0, w1b, c1_b0, Y0, 20000, 896, 640); }
  { dim3 g(512 / 64, (40000 + 127) / 128);
    k_gemm_bf16<<<g, 256, 0, stream>>>(stage1 + 640, 2432, 512, 1, w2b, nullptr, Y1, 40000, 512, 512); }
  { dim3 g(384 / 64, (40000 + 127) / 128);
    k_gemm_bf16<<<g, 256, 0, stream>>>(stage1 + 1664, 2432, 384, 1, w3b, nullptr, Y2, 40000, 384, 384); }

  k_combine1<<<E_EDGES, 256, 0, stream>>>(Y0, Y1, Y2, msgh);
  k_grid_stage2<<<E_EDGES, 64, 0, stream>>>(msgh, Y0, to_grid, from_grid, stage2);

  // SO2 conv2 (WMMA bf16)
  { dim3 g(640 / 64, (20000 + 127) / 128);
    k_gemm_bf16<<<g, 256, 0, stream>>>(stage2, 1216, 0, 0, w4b, c2_b0, Z0, 20000, 640, 320); }
  { dim3 g(1024 / 64, (40000 + 127) / 128);
    k_gemm_bf16<<<g, 256, 0, stream>>>(stage2 + 320, 1216, 256, 1, w5b, nullptr, Z1, 40000, 1024, 256); }
  { dim3 g(768 / 64, (40000 + 127) / 128);
    k_gemm_bf16<<<g, 256, 0, stream>>>(stage2 + 768, 1216, 192, 1, w6b, nullptr, Z2, 40000, 768, 192); }

  k_combine2<<<E_EDGES, 256, 0, stream>>>(Z0, Z1, Z2, msg2);

  // segmented softmax attention
  k_alpha1<<<(E_EDGES * 8 + 255) / 256, 256, 0, stream>>>(Y0, aln_g, aln_b, adot, eidx, alpha, amax);
  k_alpha2<<<(E_EDGES * 8 + 255) / 256, 256, 0, stream>>>(alpha, amax, den, eidx);

  // weight, rotate back, segment-sum
  k_attn_scatter<<<E_EDGES, 128, 0, stream>>>(msg2, wig_inv, alpha, den, eidx, node);

  // per-l output projection
  k_out_proj<<<NNODES, 256, 0, stream>>>(node, proj_w, proj_b, (float*)d_out);
}
